// ChartingAE_73392401154160
// MI455X (gfx1250) — compile-verified, matrix operand order checked by reading, not run on hardware
//
#include <hip/hip_runtime.h>
#include <hip/hip_bf16.h>

// ---------------------------------------------------------------------------
// Types for CDNA5 WMMA
// ---------------------------------------------------------------------------
typedef __attribute__((ext_vector_type(16))) __bf16        v16bf;
typedef __attribute__((ext_vector_type(8)))  float         v8f;
typedef __attribute__((ext_vector_type(8)))  unsigned int  v8u;

__device__ __forceinline__ unsigned short f2bf(float f) {
    unsigned int u = __builtin_bit_cast(unsigned int, f);
    unsigned int r = (u + 0x7fffu + ((u >> 16) & 1u)) >> 16;   // RNE
    return (unsigned short)r;
}

// ---------------------------------------------------------------------------
// f32 -> bf16 conversion (grid-stride), for activations (x)
// ---------------------------------------------------------------------------
__global__ void f2bf_kernel(const float* __restrict__ in,
                            unsigned short* __restrict__ out, long n) {
    long i = (long)blockIdx.x * blockDim.x + threadIdx.x;
    long stride = (long)gridDim.x * blockDim.x;
    for (; i < n; i += stride) out[i] = f2bf(in[i]);
}

// ---------------------------------------------------------------------------
// Fused f32->bf16 convert + transpose for weights:
//   in : [C][K][N] f32     out: [C][Npad][K] bf16 (rows n>=N zero-filled)
// LDS-tiled 32x32; weights touched exactly once (~150MB -> ~7us @ 23.3TB/s).
// ---------------------------------------------------------------------------
__global__ __launch_bounds__(256)
void cvt_transpose_kernel(const float* __restrict__ in, unsigned short* __restrict__ out,
                          int K, int N, int Npad)
{
    __shared__ float tile[32][33];
    const int c  = blockIdx.z;
    const int n0 = blockIdx.x * 32;   // over Npad
    const int k0 = blockIdx.y * 32;
    const int tx = threadIdx.x & 31;
    const int ty = threadIdx.x >> 5;  // 0..7

#pragma unroll
    for (int r = ty; r < 32; r += 8) {
        const int n = n0 + tx;
        float v = 0.f;
        if (n < N) v = in[(long)c * K * N + (long)(k0 + r) * N + n];
        tile[r][tx] = v;
    }
    __syncthreads();
#pragma unroll
    for (int r = ty; r < 32; r += 8) {
        // out[c][n0+r][k0+tx] = in[c][k0+tx][n0+r]
        out[(long)c * Npad * K + (long)(n0 + r) * K + (k0 + tx)] = f2bf(tile[tx][r]);
    }
}

// ---------------------------------------------------------------------------
// Batched bf16 GEMM, WMMA, pre-transposed weights:
//   out[c] = act(A[c] * W[c] + bias[c]),  Wt = W^T stored [Npad][K] per chart
// Block tile 128x256, BK=32, 512 threads = 16 waves (4 M x 4 N); wave tile
// 32x64 = 2x4 tiles -> 8 v_wmma_f32_16x16x32_bf16 per wave per K-step.
// Per-lane regs: 64 acc + 16 aF + 32 bF + 12 prefetch -> no spills.
// LDS double-buffered, one barrier per K-step; LDS stores for the next tile
// are placed AFTER the WMMA block so the matrix ops only wait on frag loads.
// ---------------------------------------------------------------------------
__global__ __launch_bounds__(512, 1)
void gemm_bf16_wmma(const unsigned short* __restrict__ A, long aBatchOff, int aRowStride,
                    const unsigned short* __restrict__ Wt, long wBatchOff,
                    const float* __restrict__ bias,
                    unsigned short* __restrict__ obf, long obBatchOff, int obRowStride,
                    float* __restrict__ of32, long ofBatchOff, int ofRowStride,
                    int N, int K, int applyElu)
{
    constexpr int BM = 128, BN = 256, BK = 32, PAD = 8;
    __shared__ __align__(16) unsigned short As[2][BM][BK + PAD];
    __shared__ __align__(16) unsigned short Bs[2][BN][BK + PAD];

    const int c       = blockIdx.z;
    const int rowBase = blockIdx.y * BM;
    const int colBase = blockIdx.x * BN;
    const int t       = threadIdx.x;
    const int lane    = t & 31;
    const int wave    = t >> 5;             // 0..15
    const int wm      = (wave & 3) * 32;    // wave row offset in block tile
    const int wn      = (wave >> 2) * 64;   // wave col offset in block tile
    const int half    = lane >> 4;
    const int l15     = lane & 15;

    // staging: thread t covers A row (t>>2), B rows (t>>2)+128p, cols (t&3)*8
    const int  sRow = t >> 2;               // 0..127
    const int  sCol = (t & 3) * 8;
    const unsigned short* Ab = A  + (long)c * aBatchOff + (long)(rowBase + sRow) * aRowStride + sCol;
    const unsigned short* Wb = Wt + (long)c * wBatchOff + (long)(colBase + sRow) * K          + sCol;

    const int nk = K / BK;

    v8f acc[2][4];
#pragma unroll
    for (int m = 0; m < 2; ++m)
#pragma unroll
        for (int j = 0; j < 4; ++j) acc[m][j] = v8f{};

    // ---- prologue: stage tile 0 into buffer 0 ----
    {
        *(uint4*)&As[0][sRow][sCol] = *(const uint4*)Ab;
#pragma unroll
        for (int p = 0; p < 2; ++p)
            *(uint4*)&Bs[0][sRow + p * 128][sCol] = *(const uint4*)(Wb + (long)p * 128 * K);
    }
    __syncthreads();

    for (int i = 0; i < nk; ++i) {
        const int buf = i & 1;
        const bool hasNext = (i + 1) < nk;

        // next-tile global loads first: latency hidden under the WMMAs below
        uint4 aReg, bReg[2];
        if (hasNext) {
            const int kb = (i + 1) * BK;
            aReg = *(const uint4*)(Ab + kb);
#pragma unroll
            for (int p = 0; p < 2; ++p)
                bReg[p] = *(const uint4*)(Wb + (long)p * 128 * K + kb);
        }

        // fragment loads from current buffer
        v8u aF[2], bF[4];
#pragma unroll
        for (int m = 0; m < 2; ++m) {
            const int mr = wm + m * 16 + l15;
#pragma unroll
            for (int v = 0; v < 8; ++v) {
                const int k0 = ((v < 4) ? (v * 2) : (16 + (v - 4) * 2)) + half * 8;
                aF[m][v] = *(const unsigned int*)&As[buf][mr][k0];
            }
        }
#pragma unroll
        for (int j = 0; j < 4; ++j) {
            const int nr = wn + j * 16 + l15;
#pragma unroll
            for (int v = 0; v < 8; ++v) {
                const int k0 = ((v < 4) ? (v * 2) : (16 + (v - 4) * 2)) + half * 8;
                bF[j][v] = *(const unsigned int*)&Bs[buf][nr][k0];
            }
        }

        // 8 WMMAs per wave per K-step (only wait on the frag ds_loads)
#pragma unroll
        for (int m = 0; m < 2; ++m)
#pragma unroll
            for (int j = 0; j < 4; ++j)
                acc[m][j] = __builtin_amdgcn_wmma_f32_16x16x32_bf16(
                    false, __builtin_bit_cast(v16bf, aF[m]),
                    false, __builtin_bit_cast(v16bf, bF[j]),
                    (short)0, acc[m][j], false, false);

        // stage next tile into the other buffer (loadcnt wait lands here)
        if (hasNext) {
            *(uint4*)&As[buf ^ 1][sRow][sCol] = aReg;
#pragma unroll
            for (int p = 0; p < 2; ++p)
                *(uint4*)&Bs[buf ^ 1][sRow + p * 128][sCol] = bReg[p];
        }

        __syncthreads();
    }

    // ---- epilogue: bias + ELU, dual-format store ----
#pragma unroll
    for (int m = 0; m < 2; ++m) {
#pragma unroll
        for (int j = 0; j < 4; ++j) {
            const int gCol = colBase + wn + j * 16 + l15;
            if (gCol >= N) continue;
            const float bv = bias[(long)c * N + gCol];
#pragma unroll
            for (int r = 0; r < 8; ++r) {
                const int gRow = rowBase + wm + m * 16 + half * 8 + r;
                float v = acc[m][j][r] + bv;
                if (applyElu) v = (v > 0.f) ? v : expm1f(v);
                if (obf)  obf [(long)c * obBatchOff + (long)gRow * obRowStride + gCol] = f2bf(v);
                if (of32) of32[(long)c * ofBatchOff + (long)gRow * ofRowStride + gCol] = v;
            }
        }
    }
}

// ---------------------------------------------------------------------------
// p = sparsemax(x @ Wp + bp), one wave per batch row (C == 8)
// ---------------------------------------------------------------------------
__global__ __launch_bounds__(256)
void sparsemax_kernel(const float* __restrict__ x, const float* __restrict__ Wp,
                      const float* __restrict__ bp, float* __restrict__ p,
                      int B, int D, int C)
{
    const int wave = threadIdx.x >> 5;
    const int lane = threadIdx.x & 31;
    const int b = blockIdx.x * 8 + wave;
    if (b >= B) return;

    float acc[8];
#pragma unroll
    for (int c = 0; c < 8; ++c) acc[c] = 0.f;
    for (int d = lane; d < D; d += 32) {
        const float xv = x[(long)b * D + d];
#pragma unroll
        for (int c = 0; c < 8; ++c) acc[c] += xv * Wp[(long)d * C + c];
    }
#pragma unroll
    for (int c = 0; c < 8; ++c)
        for (int off = 16; off > 0; off >>= 1)
            acc[c] += __shfl_xor(acc[c], off, 32);

    if (lane == 0) {
        float z[8], zs[8];
#pragma unroll
        for (int c = 0; c < 8; ++c) { z[c] = acc[c] + bp[c]; zs[c] = z[c]; }
        for (int i = 1; i < 8; ++i) {            // descending insertion sort
            float v = zs[i]; int j = i - 1;
            while (j >= 0 && zs[j] < v) { zs[j + 1] = zs[j]; --j; }
            zs[j + 1] = v;
        }
        float cum = 0.f, cs = 0.f; int k = 1;
        for (int i = 0; i < 8; ++i) {
            cum += zs[i];
            if (1.f + (float)(i + 1) * zs[i] > cum) { k = i + 1; cs = cum; }
        }
        const float tau = (cs - 1.f) / (float)k;
#pragma unroll
        for (int c = 0; c < 8; ++c) p[(long)b * C + c] = fmaxf(z[c] - tau, 0.f);
    }
}

// ---------------------------------------------------------------------------
// recon_x[b,d] = sum_c p[b,c] * xrec[c,b,d] ; accumulate MSE
// ---------------------------------------------------------------------------
__global__ __launch_bounds__(256)
void combine_recon(const float* __restrict__ xrec, const float* __restrict__ p,
                   const float* __restrict__ x, float* __restrict__ recon_x,
                   float* __restrict__ mse_acc, int B, int D, int C)
{
    const long idx = (long)blockIdx.x * blockDim.x + threadIdx.x;
    const long total = (long)B * D;
    float sq = 0.f;
    if (idx < total) {
        const int b = (int)(idx / D);
        float s = 0.f;
#pragma unroll
        for (int c = 0; c < 8; ++c)
            s += p[(long)b * C + c] * xrec[(long)c * B * D + idx];
        recon_x[idx] = s;
        const float diff = s - x[idx];
        sq = diff * diff;
    }
    __shared__ float red[256];
    red[threadIdx.x] = sq;
    __syncthreads();
    for (int s = 128; s > 0; s >>= 1) {
        if ((int)threadIdx.x < s) red[threadIdx.x] += red[threadIdx.x + s];
        __syncthreads();
    }
    if (threadIdx.x == 0) atomicAdd(mse_acc, red[0]);
}

// ---------------------------------------------------------------------------
// err[b,c] = sum_d (x[b,d] - xrec[c,b,d])^2, one wave per (b,c)
// ---------------------------------------------------------------------------
__global__ __launch_bounds__(256)
void recon_errors_kernel(const float* __restrict__ xrec, const float* __restrict__ x,
                         float* __restrict__ err, int B, int D, int C)
{
    const int wave = threadIdx.x >> 5;
    const int lane = threadIdx.x & 31;
    const long pair = (long)blockIdx.x * 8 + wave;
    if (pair >= (long)B * C) return;
    const int b = (int)(pair / C);
    const int c = (int)(pair % C);
    float s = 0.f;
    for (int d = lane; d < D; d += 32) {
        const float diff = x[(long)b * D + d] - xrec[(long)c * B * D + (long)b * D + d];
        s += diff * diff;
    }
    for (int off = 16; off > 0; off >>= 1) s += __shfl_xor(s, off, 32);
    if (lane == 0) err[pair] = s;
}

// ---------------------------------------------------------------------------
// per-row losses -> accumulators: [0]=recon, [1]=trans, [2..9]=sum_p, [10]=mse
// ---------------------------------------------------------------------------
__global__ __launch_bounds__(256)
void row_losses_kernel(const float* __restrict__ err, const float* __restrict__ p,
                       float* __restrict__ acc, int B, int C)
{
    const int b = blockIdx.x * blockDim.x + threadIdx.x;
    if (b >= B) return;
    float e[8], pv[8];
#pragma unroll
    for (int c = 0; c < 8; ++c) { e[c] = err[(long)b * C + c]; pv[c] = p[(long)b * C + c]; }
    float recon = 0.f;
#pragma unroll
    for (int c = 0; c < 8; ++c) recon += pv[c] * e[c];
    float m = -e[0];
#pragma unroll
    for (int c = 1; c < 8; ++c) m = fmaxf(m, -e[c]);
    float se = 0.f, q[8];
#pragma unroll
    for (int c = 0; c < 8; ++c) { q[c] = expf(-e[c] - m); se += q[c]; }
    float trans = 0.f;
#pragma unroll
    for (int c = 0; c < 8; ++c) trans -= (q[c] / se) * logf(pv[c] + 1e-8f);
    atomicAdd(acc + 0, recon);
    atomicAdd(acc + 1, trans);
#pragma unroll
    for (int c = 0; c < 8; ++c) atomicAdd(acc + 2 + c, pv[c]);
}

__global__ void finalize_kernel(const float* __restrict__ acc, float* __restrict__ outS,
                                int B, int D, int C)
{
    if (threadIdx.x == 0 && blockIdx.x == 0) {
        const float recon = acc[0] / (float)B;
        const float trans = acc[1] / (float)B;
        const float mse   = acc[10] / ((float)B * (float)D);
        float nondom = 0.f;
        for (int c = 0; c < C; ++c) {
            const float d = acc[2 + c] / (float)B - 1.f / (float)C;
            nondom += d * d;
        }
        outS[0] = recon + 1.0f * trans + 1.0f * nondom;  // total (GAMMA=BETA=1)
        outS[1] = recon;
        outS[2] = nondom;
        outS[3] = trans;
        outS[4] = mse;
    }
}

// ---------------------------------------------------------------------------
// Host orchestration
// ---------------------------------------------------------------------------
extern "C" void kernel_launch(void* const* d_in, const int* in_sizes, int n_in,
                              void* d_out, int out_size, void* d_ws, size_t ws_size,
                              hipStream_t stream) {
    constexpr int B = 4096, D = 512, L = 64, H = 1024, C = 8;
    constexpr int BN = 256;
    constexpr int Hpad = H;                 // 1024 % 256 == 0
    constexpr int Dpad = D;                 // 512  % 256 == 0
    constexpr int Lpad = BN;                // 64 -> 256 (zero-padded rows)

    const float* x   = (const float*)d_in[0];
    const float* We1 = (const float*)d_in[1];  const float* be1 = (const float*)d_in[2];
    const float* We2 = (const float*)d_in[3];  const float* be2 = (const float*)d_in[4];
    const float* We3 = (const float*)d_in[5];  const float* be3 = (const float*)d_in[6];
    const float* We4 = (const float*)d_in[7];  const float* be4 = (const float*)d_in[8];
    const float* Wd1 = (const float*)d_in[9];  const float* bd1 = (const float*)d_in[10];
    const float* Wd2 = (const float*)d_in[11]; const float* bd2 = (const float*)d_in[12];
    const float* Wd3 = (const float*)d_in[13]; const float* bd3 = (const float*)d_in[14];
    const float* Wd4 = (const float*)d_in[15]; const float* bd4 = (const float*)d_in[16];
    const float* Wp  = (const float*)d_in[17]; const float* bp  = (const float*)d_in[18];

    float* out = (float*)d_out;
    float* out_recon_x = out;                                 // [B,D]
    float* out_z       = out + (long)B * D;                   // [B,C,L]
    float* out_p       = out_z + (long)B * C * L;             // [B,C]
    float* out_scalars = out_p + (long)B * C;                 // 5 scalars

    // workspace layout
    char* base = (char*)d_ws;
    size_t off = 0;
    auto alloc = [&](size_t bytes) -> char* {
        char* p = base + off;
        off += (bytes + 255) & ~(size_t)255;
        return p;
    };
    unsigned short* xb    = (unsigned short*)alloc((size_t)B * D * 2);
    unsigned short* We1T  = (unsigned short*)alloc((size_t)C * Hpad * D * 2);  // [C][H][D]
    unsigned short* We2T  = (unsigned short*)alloc((size_t)C * Hpad * H * 2);
    unsigned short* We3T  = (unsigned short*)alloc((size_t)C * Hpad * H * 2);
    unsigned short* We4T  = (unsigned short*)alloc((size_t)C * Lpad * H * 2);  // [C][256][H]
    unsigned short* Wd1T  = (unsigned short*)alloc((size_t)C * Hpad * L * 2);  // [C][H][L]
    unsigned short* Wd2T  = (unsigned short*)alloc((size_t)C * Hpad * H * 2);
    unsigned short* Wd3T  = (unsigned short*)alloc((size_t)C * Hpad * H * 2);
    unsigned short* Wd4T  = (unsigned short*)alloc((size_t)C * Dpad * H * 2);  // [C][D][H]
    unsigned short* hA    = (unsigned short*)alloc((size_t)C * B * H * 2);
    unsigned short* hB    = (unsigned short*)alloc((size_t)C * B * H * 2);
    unsigned short* zb    = (unsigned short*)alloc((size_t)B * C * L * 2);
    float*          xrec  = (float*)alloc((size_t)C * B * D * 4);
    float*          err   = (float*)alloc((size_t)B * C * 4);
    float*          acc   = (float*)alloc(16 * 4);

    hipMemsetAsync(acc, 0, 16 * 4, stream);

    // activation conversion
    f2bf_kernel<<<2048, 256, 0, stream>>>(x, xb, (long)B * D);

    // weight convert + transpose: in [C][K][N] -> out [C][Npad][K]
    auto cvtT = [&](const float* src, unsigned short* dst, int K, int N, int Npad) {
        dim3 grid(Npad / 32, K / 32, C);
        cvt_transpose_kernel<<<grid, 256, 0, stream>>>(src, dst, K, N, Npad);
    };
    cvtT(We1, We1T, D, H, Hpad);
    cvtT(We2, We2T, H, H, Hpad);
    cvtT(We3, We3T, H, H, Hpad);
    cvtT(We4, We4T, H, L, Lpad);
    cvtT(Wd1, Wd1T, L, H, Hpad);
    cvtT(Wd2, Wd2T, H, H, Hpad);
    cvtT(Wd3, Wd3T, H, H, Hpad);
    cvtT(Wd4, Wd4T, H, D, Dpad);

    // chart probabilities (independent of GEMM chain)
    sparsemax_kernel<<<B / 8, 256, 0, stream>>>(x, Wp, bp, out_p, B, D, C);

    auto gemm = [&](const unsigned short* A, long aBO, int aRS,
                    const unsigned short* Wt, long wBO, const float* bias,
                    unsigned short* obf, long obBO, int obRS,
                    float* of32, long ofBO, int ofRS,
                    int N, int K, int elu) {
        dim3 grid((N + BN - 1) / BN, B / 128, C);
        gemm_bf16_wmma<<<grid, 512, 0, stream>>>(A, aBO, aRS, Wt, wBO, bias,
                                                 obf, obBO, obRS, of32, ofBO, ofRS,
                                                 N, K, elu);
    };

    // encoder
    gemm(xb, 0, D,           We1T, (long)Hpad * D, be1, hA, (long)B * H, H, nullptr, 0, 0, H, D, 1);
    gemm(hA, (long)B * H, H, We2T, (long)Hpad * H, be2, hB, (long)B * H, H, nullptr, 0, 0, H, H, 1);
    gemm(hB, (long)B * H, H, We3T, (long)Hpad * H, be3, hA, (long)B * H, H, nullptr, 0, 0, H, H, 1);
    gemm(hA, (long)B * H, H, We4T, (long)Lpad * H, be4, zb, L, C * L, out_z, L, C * L, L, H, 0);
    // decoder
    gemm(zb, L, C * L,       Wd1T, (long)Hpad * L, bd1, hA, (long)B * H, H, nullptr, 0, 0, H, L, 1);
    gemm(hA, (long)B * H, H, Wd2T, (long)Hpad * H, bd2, hB, (long)B * H, H, nullptr, 0, 0, H, H, 1);
    gemm(hB, (long)B * H, H, Wd3T, (long)Hpad * H, bd3, hA, (long)B * H, H, nullptr, 0, 0, H, H, 1);
    gemm(hA, (long)B * H, H, Wd4T, (long)Dpad * H, bd4, nullptr, 0, 0,
         xrec, (long)B * D, D, D, H, 0);

    // reductions / losses
    combine_recon<<<(B * D + 255) / 256, 256, 0, stream>>>(xrec, out_p, x, out_recon_x,
                                                           acc + 10, B, D, C);
    recon_errors_kernel<<<(B * C) / 8, 256, 0, stream>>>(xrec, x, err, B, D, C);
    row_losses_kernel<<<(B + 255) / 256, 256, 0, stream>>>(err, out_p, acc, B, C);
    finalize_kernel<<<1, 1, 0, stream>>>(acc, out_scalars, B, D, C);
}
